// MixtureOfExperts_73435350827097
// MI455X (gfx1250) — matrix operand order repaired and checked by reading, not compiled, not run on hardware
//
#include <hip/hip_runtime.h>
#include <hip/hip_bf16.h>

// ---------------------------------------------------------------------------
// MoE FFN for gfx1250 (MI455X): bf16 WMMA path, f32 accumulate, async-LDS
// double-buffered GEMM pipeline (explicit 2x-unrolled so buffer indices are
// compile-time constants -> no accumulator shuffling).
// out = LN( 0.25 * sum_{e in {0,1,sel0,sel1}} ( gelu(x@W1[e]+b1[e]) @ W2[e] + b2[e] ) )
// plus router logits appended to d_out.
// ---------------------------------------------------------------------------

typedef __attribute__((ext_vector_type(16))) __bf16       v16bf;
typedef __attribute__((ext_vector_type(8)))  float        v8f;
typedef __attribute__((ext_vector_type(4)))  float        v4f;
typedef __attribute__((ext_vector_type(4)))  unsigned int v4u;
typedef __attribute__((ext_vector_type(2)))  unsigned int v2u;

#define T_TOK   4096   // B*S
#define DMODEL  1024
#define DFF     4096
#define NEXP    6
#define BM      128
#define BN      128
#define BK      32
#define LDST    40     // LDS row stride in bf16 elements (80 B, 16B-aligned)

union FragU  { v16bf v; v4u q[2]; };
union Pack4  { __bf16 b[4]; v2u u2; };

__device__ __forceinline__ float gelu_exact(float x) {
    return 0.5f * x * (1.0f + erff(x * 0.70710678118654752440f));
}

__device__ __forceinline__ unsigned lds_off32(const void* p) {
    // AS3 pointer -> flat: low 32 bits are the LDS byte offset.
    return (unsigned)(unsigned long long)p;
}

// CDNA5 async copy: 16 bytes per lane, global -> LDS, tracked by ASYNCcnt.
__device__ __forceinline__ void async_b128(unsigned lds_byte, unsigned gbyte_off,
                                           const __bf16* sbase) {
    asm volatile("global_load_async_to_lds_b128 %0, %1, %2"
                 :: "v"(lds_byte), "v"(gbyte_off), "s"(sbase)
                 : "memory");
}

// Stage one 128x32 bf16 tile (row stride K elems in global) into LDS at
// lds_base (row stride LDST elems). 2 async instructions per wave.
__device__ __forceinline__ void stage_tile(unsigned lds_base, const __bf16* g_row0,
                                           int K, int tid) {
    #pragma unroll
    for (int i = 0; i < 2; ++i) {
        int idx = i * 256 + tid;               // 0..511, lane-coalesced 16B chunks
        int r = idx >> 2, c8 = (idx & 3) * 8;
        unsigned lds = lds_base + (unsigned)(r * LDST + c8) * 2u;
        unsigned gof = (unsigned)(r * K + c8) * 2u;
        async_b128(lds, gof, g_row0);
    }
}

// ---------------------------------------------------------------------------
// Core compute for one K-step from LDS buffers (8 waves, 128x128 tile).
// ---------------------------------------------------------------------------
__device__ __forceinline__ void wmma_step(const __bf16* As, const __bf16* Bs,
                                          v8f acc[4][2], int wm, int wn,
                                          int h, int l16) {
    FragU b[2];
    #pragma unroll
    for (int j = 0; j < 2; ++j) {
        const __bf16* p = &Bs[(wn * 32 + j * 16 + l16) * LDST + h * 16];
        b[j].q[0] = *(const v4u*)(p);
        b[j].q[1] = *(const v4u*)(p + 8);
    }
    #pragma unroll
    for (int i = 0; i < 4; ++i) {
        FragU a;
        const __bf16* p = &As[(wm * 64 + i * 16 + l16) * LDST];
        a.q[0] = *(const v4u*)(p + h * 16);
        a.q[1] = *(const v4u*)(p + h * 16 + 8);
        #pragma unroll
        for (int j = 0; j < 2; ++j)
            acc[i][j] = __builtin_amdgcn_wmma_f32_16x16x32_bf16(
                false, a.v, false, b[j].v, (short)0, acc[i][j], false, false);
    }
}

// ---------------------------------------------------------------------------
// Double-buffered async main loop, explicitly unrolled by 2 (nit is even).
// Per half-step: issue 4 async copies for next buffer, s_wait_asynccnt 0x4
// (async loads complete in order -> the 4 oldest = current buffer are done),
// barrier, compute, barrier.
// ---------------------------------------------------------------------------
__device__ __forceinline__ void gemm_mainloop(const __bf16* Ag, const __bf16* Bg,
                                              int K,
                                              __bf16 (*As)[BM * LDST],
                                              __bf16 (*Bs)[BN * LDST],
                                              v8f acc[4][2],
                                              int tid, int wm, int wn,
                                              int h, int l16) {
    const unsigned as0 = lds_off32(&As[0][0]), as1 = lds_off32(&As[1][0]);
    const unsigned bs0 = lds_off32(&Bs[0][0]), bs1 = lds_off32(&Bs[1][0]);

    stage_tile(as0, Ag, K, tid);
    stage_tile(bs0, Bg, K, tid);

    const int nit = K / BK;                    // even for all our shapes
    for (int it = 0; it < nit; it += 2) {
        // even half: compute buf0, prefetch buf1 (it+1 always < nit)
        stage_tile(as1, Ag + (it + 1) * BK, K, tid);
        stage_tile(bs1, Bg + (it + 1) * BK, K, tid);
        asm volatile("s_wait_asynccnt 0x4" ::: "memory");
        __syncthreads();
        wmma_step(As[0], Bs[0], acc, wm, wn, h, l16);
        __syncthreads();

        // odd half: compute buf1, prefetch buf0 if more work remains
        if (it + 2 < nit) {
            stage_tile(as0, Ag + (it + 2) * BK, K, tid);
            stage_tile(bs0, Bg + (it + 2) * BK, K, tid);
            asm volatile("s_wait_asynccnt 0x4" ::: "memory");
        } else {
            asm volatile("s_wait_asynccnt 0x0" ::: "memory");
        }
        __syncthreads();
        wmma_step(As[1], Bs[1], acc, wm, wn, h, l16);
        __syncthreads();
    }
}

// ---------------------------------------------------------------------------
// Pre-pass: f32 -> bf16 straight convert (x).
// ---------------------------------------------------------------------------
__launch_bounds__(256)
__global__ void convert_bf16_kernel(const float* __restrict__ in,
                                    __bf16* __restrict__ out) {
    int i = (blockIdx.x * 256 + threadIdx.x) * 4;
    v4f f = *(const v4f*)(in + i);
    Pack4 p;
    p.b[0] = (__bf16)f.x; p.b[1] = (__bf16)f.y;
    p.b[2] = (__bf16)f.z; p.b[3] = (__bf16)f.w;
    *(v2u*)(out + i) = p.u2;
}

// ---------------------------------------------------------------------------
// Pre-pass: transpose + convert. in: R x C f32 (row-major) -> out: C x R bf16.
// ---------------------------------------------------------------------------
__launch_bounds__(256)
__global__ void transpose_convert_kernel(const float* __restrict__ in,
                                         __bf16* __restrict__ out,
                                         int R, int C) {
    __shared__ float tile[32][33];
    const int bc = blockIdx.x * 32, br = blockIdx.y * 32;
    const int tx = threadIdx.x & 31, ty = threadIdx.x >> 5;
    #pragma unroll
    for (int i = 0; i < 4; ++i) {
        int r = ty + i * 8;
        tile[r][tx] = in[(size_t)(br + r) * C + bc + tx];
    }
    __syncthreads();
    #pragma unroll
    for (int i = 0; i < 4; ++i) {
        int r = ty + i * 8;                       // output row = input col
        out[(size_t)(bc + r) * R + br + tx] = (__bf16)tile[tx][r];
    }
}

// ---------------------------------------------------------------------------
// Router: logits = x @ router_W (1024x4); top-2 indices (+FIXED offset) to ws.
// ---------------------------------------------------------------------------
__launch_bounds__(256)
__global__ void moe_router_kernel(const float* __restrict__ X,
                                  const float* __restrict__ RW,
                                  float* __restrict__ logits,
                                  int2* __restrict__ sel)
{
    int t = blockIdx.x * blockDim.x + threadIdx.x;
    if (t >= T_TOK) return;
    const float* x = X + (size_t)t * DMODEL;
    float l0 = 0.f, l1 = 0.f, l2 = 0.f, l3 = 0.f;
    for (int d = 0; d < DMODEL; d += 4) {
        v4f xv = *(const v4f*)(x + d);
        const float* w = RW + (size_t)d * 4;
        l0 += xv.x * w[0]  + xv.y * w[4]  + xv.z * w[8]  + xv.w * w[12];
        l1 += xv.x * w[1]  + xv.y * w[5]  + xv.z * w[9]  + xv.w * w[13];
        l2 += xv.x * w[2]  + xv.y * w[6]  + xv.z * w[10] + xv.w * w[14];
        l3 += xv.x * w[3]  + xv.y * w[7]  + xv.z * w[11] + xv.w * w[15];
    }
    float l[4] = { l0, l1, l2, l3 };
    float* lo = logits + (size_t)t * 4;
    lo[0] = l0; lo[1] = l1; lo[2] = l2; lo[3] = l3;
    int i0 = 0; float v0 = l[0];
    #pragma unroll
    for (int e = 1; e < 4; ++e) if (l[e] > v0) { v0 = l[e]; i0 = e; }
    int i1 = -1; float v1 = -3.402823466e38f;
    #pragma unroll
    for (int e = 0; e < 4; ++e) if (e != i0 && l[e] > v1) { v1 = l[e]; i1 = e; }
    sel[t] = make_int2(i0 + 2, i1 + 2);
}

// ---------------------------------------------------------------------------
// GEMM1 + bias + exact GELU: H(bf16,TxF) = gelu( Xb(bf16,TxD) @ W1t^T + b1 )
// W1t is N-major (F x D bf16).
// ---------------------------------------------------------------------------
__launch_bounds__(256)
__global__ void moe_gemm1_gelu_kernel(const __bf16* __restrict__ Xb,
                                      const __bf16* __restrict__ W1t,
                                      const float* __restrict__ b1,
                                      __bf16* __restrict__ H)
{
    __shared__ __align__(16) __bf16 As[2][BM * LDST];
    __shared__ __align__(16) __bf16 Bs[2][BN * LDST];

    const int tid  = threadIdx.x;
    const int lane = tid & 31;
    const int wave = tid >> 5;
    const int wm   = wave >> 2, wn = wave & 3;
    const int bm   = blockIdx.y, bn = blockIdx.x;
    const int h    = lane >> 4, l16 = lane & 15;

    v8f acc[4][2];
    #pragma unroll
    for (int i = 0; i < 4; ++i)
        #pragma unroll
        for (int j = 0; j < 2; ++j)
            acc[i][j] = (v8f){0.f,0.f,0.f,0.f,0.f,0.f,0.f,0.f};

    gemm_mainloop(Xb  + (size_t)(bm * BM) * DMODEL,
                  W1t + (size_t)(bn * BN) * DMODEL,
                  DMODEL, As, Bs, acc, tid, wm, wn, h, l16);

    #pragma unroll
    for (int j = 0; j < 2; ++j) {
        int col   = bn * BN + wn * 32 + j * 16 + l16;
        float bias = b1[col];
        #pragma unroll
        for (int i = 0; i < 4; ++i) {
            int row0 = bm * BM + wm * 64 + i * 16 + 8 * h;
            #pragma unroll
            for (int v = 0; v < 8; ++v) {
                float xv = acc[i][j][v] + bias;
                H[(size_t)(row0 + v) * DFF + col] = (__bf16)gelu_exact(xv);
            }
        }
    }
}

// ---------------------------------------------------------------------------
// GEMM2 + masked accumulate: OUT += mask(tok,e) * ( H @ W2t^T + b2 )
// W2t is N-major (D x F bf16). mask = 0.25 for fixed/selected experts else 0.
// ---------------------------------------------------------------------------
__launch_bounds__(256)
__global__ void moe_gemm2_combine_kernel(const __bf16* __restrict__ H,
                                         const __bf16* __restrict__ W2t,
                                         const float* __restrict__ b2,
                                         const int2* __restrict__ sel,
                                         float* __restrict__ OUT,
                                         int expert, int init)
{
    __shared__ __align__(16) __bf16 As[2][BM * LDST];
    __shared__ __align__(16) __bf16 Bs[2][BN * LDST];
    __shared__ float scales[BM];

    const int tid  = threadIdx.x;
    const int lane = tid & 31;
    const int wave = tid >> 5;
    const int wm   = wave >> 2, wn = wave & 3;
    const int bm   = blockIdx.y, bn = blockIdx.x;
    const int h    = lane >> 4, l16 = lane & 15;

    if (tid < BM) {
        int tok = bm * BM + tid;
        float s = 0.25f;
        if (expert >= 2) {
            int2 sl = sel[tok];
            s = (sl.x == expert || sl.y == expert) ? 0.25f : 0.0f;
        }
        scales[tid] = s;
    }

    v8f acc[4][2];
    #pragma unroll
    for (int i = 0; i < 4; ++i)
        #pragma unroll
        for (int j = 0; j < 2; ++j)
            acc[i][j] = (v8f){0.f,0.f,0.f,0.f,0.f,0.f,0.f,0.f};

    gemm_mainloop(H   + (size_t)(bm * BM) * DFF,
                  W2t + (size_t)(bn * BN) * DFF,
                  DFF, As, Bs, acc, tid, wm, wn, h, l16);

    #pragma unroll
    for (int j = 0; j < 2; ++j) {
        int col   = bn * BN + wn * 32 + j * 16 + l16;
        float bias = b2[col];
        #pragma unroll
        for (int i = 0; i < 4; ++i) {
            int rl0 = wm * 64 + i * 16 + 8 * h;
            #pragma unroll
            for (int v = 0; v < 8; ++v) {
                int rl = rl0 + v;
                float s = scales[rl];
                size_t o = (size_t)(bm * BM + rl) * DMODEL + col;
                float prev = init ? 0.0f : OUT[o];
                OUT[o] = prev + s * (acc[i][j][v] + bias);
            }
        }
    }
}

// ---------------------------------------------------------------------------
// LayerNorm in place over d_out's first T_TOK*DMODEL region. One WG per token.
// ---------------------------------------------------------------------------
__launch_bounds__(256)
__global__ void moe_layernorm_kernel(float* __restrict__ OUT,
                                     const float* __restrict__ gamma,
                                     const float* __restrict__ beta)
{
    __shared__ float s1[256];
    __shared__ float s2[256];
    const int t = blockIdx.x;
    const int tid = threadIdx.x;
    float* row = OUT + (size_t)t * DMODEL;

    float vals[4];
    float a = 0.f, bsq = 0.f;
    #pragma unroll
    for (int i = 0; i < 4; ++i) {
        float v = row[tid + i * 256];
        vals[i] = v;
        a += v; bsq += v * v;
    }
    s1[tid] = a; s2[tid] = bsq;
    __syncthreads();
    for (int off = 128; off > 0; off >>= 1) {
        if (tid < off) { s1[tid] += s1[tid + off]; s2[tid] += s2[tid + off]; }
        __syncthreads();
    }
    float mu  = s1[0] * (1.0f / DMODEL);
    float var = s2[0] * (1.0f / DMODEL) - mu * mu;
    float inv = rsqrtf(var + 1e-5f);
    #pragma unroll
    for (int i = 0; i < 4; ++i) {
        int c = tid + i * 256;
        row[c] = (vals[i] - mu) * inv * gamma[c] + beta[c];
    }
}

// ---------------------------------------------------------------------------
extern "C" void kernel_launch(void* const* d_in, const int* in_sizes, int n_in,
                              void* d_out, int out_size, void* d_ws, size_t ws_size,
                              hipStream_t stream) {
    (void)in_sizes; (void)n_in; (void)out_size; (void)ws_size;
    const float* x        = (const float*)d_in[0];
    const float* router_W = (const float*)d_in[1];
    const float* W1       = (const float*)d_in[2];
    const float* b1       = (const float*)d_in[3];
    const float* W2       = (const float*)d_in[4];
    const float* b2       = (const float*)d_in[5];
    const float* ln_g     = (const float*)d_in[6];
    const float* ln_b     = (const float*)d_in[7];

    float* out    = (float*)d_out;                       // (B,S,D) f32
    float* logits = out + (size_t)T_TOK * DMODEL;        // (B,S,4) f32 appended

    // workspace layout (bytes):
    //   [0, 32K)           : top-2 expert ids (int2 per token)
    //   [64K,  +8M)        : x in bf16            (T x D)
    //   [+8M,  +8M)        : W1t for cur expert   (F x D bf16, N-major)
    //   [+16M, +8M)        : W2t for cur expert   (D x F bf16, N-major)
    //   [+24M, +32M)       : H                    (T x F bf16)
    char* ws = (char*)d_ws;
    int2*   sel = (int2*)ws;
    __bf16* Xb  = (__bf16*)(ws + (1u << 16));
    __bf16* W1t = (__bf16*)(ws + (1u << 16) + (8u << 20));
    __bf16* W2t = (__bf16*)(ws + (1u << 16) + (16u << 20));
    __bf16* H   = (__bf16*)(ws + (1u << 16) + (24u << 20));

    moe_router_kernel<<<T_TOK / 256, 256, 0, stream>>>(x, router_W, logits, sel);

    convert_bf16_kernel<<<(T_TOK * DMODEL) / 1024, 256, 0, stream>>>(x, Xb);

    for (int e = 0; e < NEXP; ++e) {
        // W1[e]: (D x F) f32 -> W1t: (F x D) bf16
        transpose_convert_kernel<<<dim3(DFF / 32, DMODEL / 32), 256, 0, stream>>>(
            W1 + (size_t)e * DMODEL * DFF, W1t, DMODEL, DFF);
        moe_gemm1_gelu_kernel<<<dim3(DFF / BN, T_TOK / BM), 256, 0, stream>>>(
            Xb, W1t, b1 + (size_t)e * DFF, H);
        // W2[e]: (F x D) f32 -> W2t: (D x F) bf16
        transpose_convert_kernel<<<dim3(DMODEL / 32, DFF / 32), 256, 0, stream>>>(
            W2 + (size_t)e * DFF * DMODEL, W2t, DFF, DMODEL);
        moe_gemm2_combine_kernel<<<dim3(DMODEL / BN, T_TOK / BM), 256, 0, stream>>>(
            H, W2t, b2 + (size_t)e * DMODEL, sel, out, e, e == 0 ? 1 : 0);
    }

    moe_layernorm_kernel<<<T_TOK, 256, 0, stream>>>(out, ln_g, ln_b);
}
